// GraphConvolution_91173565759514
// MI455X (gfx1250) — matrix operand order repaired and checked by reading, not compiled, not run on hardware
//
#include <hip/hip_runtime.h>
#include <hip/hip_bf16.h>
#include <hip/hip_fp16.h>

#define N_NODES 100000
#define N_EDGES 3200000
#define F_IN 512
#define F_OUT 512

// GEMM tiling
#define BM 128
#define BN 64
#define KT 32
#define LDT 40   // padded LDS row stride in f16 elements (80 B, 16B-aligned)

typedef __attribute__((ext_vector_type(16))) _Float16 v16h;
typedef __attribute__((ext_vector_type(8)))  float    v8f;

// ---------------------------------------------------------------------------
// Fragment loaders from row-major f16 LDS tiles (CDNA5 ISA 7.12.2 layouts).
// A 16x32 f16: lane<16 holds M=lane, elems 0..7 -> K=0..7, 8..15 -> K=16..23;
//              lane>=16 shifts K by +8.
__device__ inline v16h load_frag_a(const _Float16* tile, int lane) {
    const int r  = lane & 15;
    const int hi = lane >> 4;
    const _Float16* p = tile + r * LDT + hi * 8;
    union { v16h v; uint4 q[2]; } u;
    u.q[0] = *(const uint4*)(p);
    u.q[1] = *(const uint4*)(p + 16);
    return u.v;
}
// B 32x16 f16 (tile stored N-major [n][k]): lane<16 holds N=lane, elems 0..7 -> K=0..7,
// 8..15 -> K=16..23; lane>=16 holds N=lane-16, K shifted by +8.
__device__ inline v16h load_frag_b(const _Float16* tile, int lane) {
    const int n  = lane & 15;
    const int hi = lane >> 4;
    const _Float16* p = tile + n * LDT + hi * 8;
    union { v16h v; uint4 q[2]; } u;
    u.q[0] = *(const uint4*)(p);
    u.q[1] = *(const uint4*)(p + 16);
    return u.v;
}

// ---------------------------------------------------------------------------
// Kernel 1: support[f16] = x @ W^T + b   (WMMA f16 -> f32 accumulate)
// Block = 256 threads = 8 waves; block tile BM(128) x BN(64); wave tile 32x32.
// Software-pipelined: next K-tile global loads issue while current LDS tile
// feeds ds_load_b128 + v_wmma. Boundary rows are CLAMPED (not predicated) so
// the 6 tile loads stay branch-free and batch under one s_wait_loadcnt;
// clamped rows only pollute C rows that the epilogue never stores.
__global__ __launch_bounds__(256)
void gcn_gemm_kernel(const float* __restrict__ x, const float* __restrict__ W,
                     const float* __restrict__ b, _Float16* __restrict__ sup) {
    __shared__ _Float16 Alds[BM * LDT];
    __shared__ _Float16 Blds[BN * LDT];

    const int tid  = threadIdx.x;
    const int lane = tid & 31;
    const int wave = tid >> 5;
    const int wm   = wave & 3;     // M offset = wm*32
    const int wn   = wave >> 2;    // N offset = wn*32
    const int gm0  = blockIdx.y * BM;
    const int gn0  = blockIdx.x * BN;

    v8f acc[2][2];
    v8f zf = {};
    acc[0][0] = zf; acc[0][1] = zf; acc[1][0] = zf; acc[1][1] = zf;

    float4 aR[4];
    float4 bR[2];

    // per-thread tile coordinates (reused every iteration)
    int arow[4], ac4[4];
    #pragma unroll
    for (int c = 0; c < 4; ++c) {
        int q = tid + 256 * c;
        arow[c] = q >> 3;
        ac4[c]  = q & 7;
    }

    auto load_tiles = [&](int kt) {
        #pragma unroll
        for (int c = 0; c < 4; ++c) {
            int gr = gm0 + arow[c];
            if (gr > N_NODES - 1) gr = N_NODES - 1;   // clamp: branch-free
            aR[c] = *(const float4*)(x + (size_t)gr * F_IN + kt + ac4[c] * 4);
        }
        #pragma unroll
        for (int c = 0; c < 2; ++c) {
            bR[c] = *(const float4*)(W + (size_t)(gn0 + arow[c]) * F_IN + kt + ac4[c] * 4);
        }
    };
    auto store_tiles = [&]() {
        #pragma unroll
        for (int c = 0; c < 4; ++c) {
            union { _Float16 h[4]; unsigned long long u64; } pk;
            pk.h[0] = (_Float16)aR[c].x; pk.h[1] = (_Float16)aR[c].y;
            pk.h[2] = (_Float16)aR[c].z; pk.h[3] = (_Float16)aR[c].w;
            *(unsigned long long*)(&Alds[arow[c] * LDT + ac4[c] * 4]) = pk.u64;
        }
        #pragma unroll
        for (int c = 0; c < 2; ++c) {
            union { _Float16 h[4]; unsigned long long u64; } pk;
            pk.h[0] = (_Float16)bR[c].x; pk.h[1] = (_Float16)bR[c].y;
            pk.h[2] = (_Float16)bR[c].z; pk.h[3] = (_Float16)bR[c].w;
            *(unsigned long long*)(&Blds[arow[c] * LDT + ac4[c] * 4]) = pk.u64;
        }
    };

    load_tiles(0);   // prologue

    for (int kt = 0; kt < F_IN; kt += KT) {
        store_tiles();
        __syncthreads();

        // issue next tile's global loads; they overlap the LDS reads + WMMAs
        if (kt + KT < F_IN) load_tiles(kt + KT);

        v16h a0 = load_frag_a(Alds + (wm * 32 +  0) * LDT, lane);
        v16h a1 = load_frag_a(Alds + (wm * 32 + 16) * LDT, lane);
        v16h b0 = load_frag_b(Blds + (wn * 32 +  0) * LDT, lane);
        v16h b1 = load_frag_b(Blds + (wn * 32 + 16) * LDT, lane);

        acc[0][0] = __builtin_amdgcn_wmma_f32_16x16x32_f16(false, a0, false, b0,
                        (short)0, acc[0][0], false, false);
        acc[0][1] = __builtin_amdgcn_wmma_f32_16x16x32_f16(false, a0, false, b1,
                        (short)0, acc[0][1], false, false);
        acc[1][0] = __builtin_amdgcn_wmma_f32_16x16x32_f16(false, a1, false, b0,
                        (short)0, acc[1][0], false, false);
        acc[1][1] = __builtin_amdgcn_wmma_f32_16x16x32_f16(false, a1, false, b1,
                        (short)0, acc[1][1], false, false);
        __syncthreads();
    }

    // --- epilogue: +bias, f32 -> f16 store (C/D layout: lane->N, vgpr->M)
    const int nlo = lane & 15;
    const int mhi = (lane >> 4) << 3;
    #pragma unroll
    for (int j = 0; j < 2; ++j) {
        const int colg = gn0 + wn * 32 + j * 16 + nlo;
        const float bias = b[colg];
        #pragma unroll
        for (int i = 0; i < 2; ++i) {
            const int mbase = gm0 + wm * 32 + i * 16 + mhi;
            #pragma unroll
            for (int v = 0; v < 8; ++v) {
                const int m = mbase + v;
                if (m < N_NODES)
                    sup[(size_t)m * F_OUT + colg] = (_Float16)(acc[i][j][v] + bias);
            }
        }
    }
}

// ---------------------------------------------------------------------------
// CSR-build helpers
__global__ void zero_int_kernel(int* __restrict__ p, int n) {
    int i = blockIdx.x * blockDim.x + threadIdx.x;
    if (i < n) p[i] = 0;
}

__global__ void count_kernel(const int* __restrict__ rows, int* __restrict__ cnt, int n) {
    int i = blockIdx.x * blockDim.x + threadIdx.x;
    if (i < n) atomicAdd(&cnt[rows[i]], 1);
}

// Single-block exclusive scan over N counts -> row_ptr[N+1] and cursor copy.
__global__ __launch_bounds__(1024)
void scan_kernel(const int* __restrict__ cnt, int* __restrict__ row_ptr,
                 int* __restrict__ cursor, int n) {
    __shared__ int sdata[1024];
    __shared__ int carry;
    const int tid = threadIdx.x;
    if (tid == 0) carry = 0;
    __syncthreads();
    for (int base = 0; base < n; base += 1024) {
        const int i = base + tid;
        const int v = (i < n) ? cnt[i] : 0;
        sdata[tid] = v;
        __syncthreads();
        for (int off = 1; off < 1024; off <<= 1) {
            int t = (tid >= off) ? sdata[tid - off] : 0;
            __syncthreads();
            sdata[tid] += t;
            __syncthreads();
        }
        const int c0   = carry;
        const int excl = c0 + sdata[tid] - v;
        if (i < n) { row_ptr[i] = excl; cursor[i] = excl; }
        __syncthreads();
        if (tid == 0) carry = c0 + sdata[1023];
        __syncthreads();
    }
    if (tid == 0) row_ptr[n] = carry;
}

__global__ void scatter_kernel(const int* __restrict__ rows, int* __restrict__ cursor,
                               int* __restrict__ perm, int n) {
    int i = blockIdx.x * blockDim.x + threadIdx.x;
    if (i < n) {
        int pos = atomicAdd(&cursor[rows[i]], 1);
        perm[pos] = i;
    }
}

// ---------------------------------------------------------------------------
// Kernel 2: out[row,:] = sum_{e in row} vals[e] * support[cols[e], :]
// One wave32 per output row; lane owns 16 consecutive f32 features.
__global__ __launch_bounds__(256)
void gather_kernel(const _Float16* __restrict__ sup, const int* __restrict__ row_ptr,
                   const int* __restrict__ perm, const int* __restrict__ cols,
                   const float* __restrict__ vals, float* __restrict__ out) {
    const int wave = (blockIdx.x * blockDim.x + threadIdx.x) >> 5;
    const int lane = threadIdx.x & 31;
    if (wave >= N_NODES) return;

    const int beg = row_ptr[wave];
    const int end = row_ptr[wave + 1];
    const int fbase = lane * 16;

    float acc[16];
    #pragma unroll
    for (int j = 0; j < 16; ++j) acc[j] = 0.f;

    int idx = (beg < end) ? perm[beg] : 0;
    for (int e = beg; e < end; ++e) {
        const int idx_next = (e + 1 < end) ? perm[e + 1] : 0;  // pipeline perm fetch
        const int   c = cols[idx];
        const float v = vals[idx];
        const _Float16* p = sup + (size_t)c * F_OUT + fbase;
        union { uint4 q; _Float16 h[8]; } u0, u1;
        u0.q = *(const uint4*)(p);
        u1.q = *(const uint4*)(p + 8);
        #pragma unroll
        for (int j = 0; j < 8; ++j) acc[j]     += v * (float)u0.h[j];
        #pragma unroll
        for (int j = 0; j < 8; ++j) acc[8 + j] += v * (float)u1.h[j];
        idx = idx_next;
    }

    float* op = out + (size_t)wave * F_OUT + fbase;
    #pragma unroll
    for (int j = 0; j < 4; ++j)
        *(float4*)(op + 4 * j) = make_float4(acc[4 * j], acc[4 * j + 1],
                                             acc[4 * j + 2], acc[4 * j + 3]);
}

// ---------------------------------------------------------------------------
extern "C" void kernel_launch(void* const* d_in, const int* in_sizes, int n_in,
                              void* d_out, int out_size, void* d_ws, size_t ws_size,
                              hipStream_t stream) {
    const float* x    = (const float*)d_in[0];
    const float* W    = (const float*)d_in[1];
    const float* b    = (const float*)d_in[2];
    const int*   rows = (const int*)d_in[3];
    const int*   cols = (const int*)d_in[4];
    const float* vals = (const float*)d_in[5];
    float*       out  = (float*)d_out;

    // workspace layout (all 16B aligned)
    char* ws = (char*)d_ws;
    _Float16* sup = (_Float16*)ws;
    size_t off = (size_t)N_NODES * F_OUT * sizeof(_Float16);   // 102,400,000
    int* cnt = (int*)(ws + off);      off += (size_t)N_NODES * 4;        // 400,000
    int* row_ptr = (int*)(ws + off);  off += (size_t)(N_NODES + 1) * 4 + 12; // pad->400,016
    int* cursor = (int*)(ws + off);   off += (size_t)N_NODES * 4;        // 400,000
    int* perm = (int*)(ws + off);                                        // 12,800,000

    // 1) zero edge counts
    zero_int_kernel<<<(N_NODES + 255) / 256, 256, 0, stream>>>(cnt, N_NODES);
    // 2) support = x @ W^T + b  (WMMA f16->f32, stored f16)
    dim3 ggrid(F_OUT / BN, (N_NODES + BM - 1) / BM);
    gcn_gemm_kernel<<<ggrid, 256, 0, stream>>>(x, W, b, sup);
    // 3) CSR build: count / scan / scatter (int atomics only)
    count_kernel<<<(N_EDGES + 255) / 256, 256, 0, stream>>>(rows, cnt, N_EDGES);
    scan_kernel<<<1, 1024, 0, stream>>>(cnt, row_ptr, cursor, N_NODES);
    scatter_kernel<<<(N_EDGES + 255) / 256, 256, 0, stream>>>(rows, cursor, perm, N_EDGES);
    // 4) gather SpMM: one wave per row, no float atomics, full coverage of out
    gather_kernel<<<(N_NODES * 32 + 255) / 256, 256, 0, stream>>>(sup, row_ptr, perm,
                                                                  cols, vals, out);
}
